// LSTMNet_34943853920749
// MI455X (gfx1250) — matrix operand order repaired and checked by reading, not compiled, not run on hardware
//
#include <hip/hip_runtime.h>
#include <hip/hip_bf16.h>
#include <stdint.h>

typedef __attribute__((ext_vector_type(8)))  float    v8f;
typedef __attribute__((ext_vector_type(8)))  __bf16   v8bf;
typedef __attribute__((ext_vector_type(16))) __bf16   v16bf;
typedef __attribute__((ext_vector_type(4)))  unsigned u32x4;
typedef __attribute__((ext_vector_type(4)))  int      i32x4;
typedef __attribute__((ext_vector_type(8)))  int      i32x8;

#define B_   32
#define T_   2048
#define DIN  128
#define H_   512
#define G4   2048          // 4*H
#define O_   128
#define K0   640           // DIN + H
#define K1   1024          // H + H
#define NWG  32            // one WG per 16 h-columns
#define HC   16            // h columns per WG

// ---- workspace layout (bytes, 256-aligned) ----
#define X_OFF      0ull
#define X_BYTES    ((unsigned long long)B_*T_*DIN*2ull)      // 16 MB bf16 x
#define W0_OFF     (X_OFF + X_BYTES)
#define W0_BYTES   ((unsigned long long)G4*K0*2ull)          // [2048][640]  bf16 [w_ih0|w_hh0]
#define W1_OFF     (W0_OFF + W0_BYTES)
#define W1_BYTES   ((unsigned long long)G4*K1*2ull)          // [2048][1024] bf16 [w_ih1|w_hh1]
#define FC_OFF     (W1_OFF + W1_BYTES)
#define FC_BYTES   ((unsigned long long)O_*H_*2ull)
#define H0_OFF     (FC_OFF + FC_BYTES)
#define HP_BYTES   (2ull*B_*H_*2ull)                         // double-buffered h publish, bf16
#define H1_OFF     (H0_OFF + HP_BYTES)
#define OUT1_OFF   (H1_OFF + HP_BYTES)
#define OUT1_BYTES ((unsigned long long)B_*T_*H_*2ull)       // 64 MB bf16 layer-1 hidden stream
#define BAR_OFF    (OUT1_OFF + OUT1_BYTES)

// ---- d_out regions (float elements) ----
#define OUT_ELEMS  (B_*T_*O_)        // 8388608
#define HID_ELEMS  (2*B_*O_)         // 8192
#define CN_OFF_EL  (OUT_ELEMS + HID_ELEMS)

// ---- dynamic LDS layout (bytes) ----
#define LDS_W0     0u
#define LDS_W0_B   (64u*K0*2u)       // 80 KB  weight slice layer0 [64][640]
#define LDS_W1     (LDS_W0 + LDS_W0_B)
#define LDS_W1_B   (64u*K1*2u)       // 128 KB weight slice layer1 [64][1024]
#define LDS_G      (LDS_W1 + LDS_W1_B)
#define LDS_G_B    (4u*32u*16u*4u)   // gate accumulators f32 [4][32][16]
#define LDS_C0     (LDS_G + LDS_G_B)
#define LDS_C_B    (32u*16u*4u)      // cell state slice f32 [32][16]
#define LDS_C1     (LDS_C0 + LDS_C_B)
#define LDS_BS     (LDS_C1 + LDS_C_B)
#define LDS_BS_B   (2u*4u*16u*4u)    // combined biases [layer][gate][16]
#define SMEM_BYTES (LDS_BS + LDS_BS_B)   // 225,792 B < 320 KB WGP LDS

// ---------------- helpers ----------------
__device__ __forceinline__ float sigm(float x) { return 1.f / (1.f + __expf(-x)); }
__device__ __forceinline__ float tanh_f(float x) {
    if (x >  10.f) return  1.f;
    if (x < -10.f) return -1.f;
    float e = __expf(2.f * x);
    return (e - 1.f) / (e + 1.f);
}

// A-fragment (16x32 bf16, ISA 7.12.2): lane<16 rows hold K {0..7,16..23}; lanes>=16 hold K {8..15,24..31}
__device__ __forceinline__ v16bf load_a16(const __bf16* p, int lane) {
    const int o = (lane & 16) ? 8 : 0;
    v8bf lo = *(const v8bf*)(p + o);
    v8bf hi = *(const v8bf*)(p + o + 16);
    return __builtin_shufflevector(lo, hi, 0,1,2,3,4,5,6,7,8,9,10,11,12,13,14,15);
}

__device__ __forceinline__ v16bf relu16(v16bf a) {
#pragma unroll
    for (int e = 0; e < 16; ++e) {
        float f = (float)a[e];
        a[e] = (__bf16)(f > 0.f ? f : 0.f);
    }
    return a;
}

// TDM: 2D tile [rows x k_elems] of bf16, row stride k_elems, global -> LDS
__device__ __forceinline__ void tdm_load_2d(unsigned lds_off, const void* gsrc, int rows, int k) {
    unsigned long long ga = (unsigned long long)(uintptr_t)gsrc;
    u32x4 g0;
    g0.x = 1u;                                             // count=1, user descriptor
    g0.y = lds_off;                                        // lds_addr (bytes)
    g0.z = (unsigned)(ga & 0xFFFFFFFFull);                 // global_addr lo
    g0.w = (unsigned)((ga >> 32) & 0x1FFFFFFull) | 0x80000000u; // global_addr hi | type=2
    i32x8 g1;
    g1[0] = 0x00010000;                                    // data_size=1 (2 bytes/elem)
    g1[1] = (k & 0xFFFF) << 16;                            // tensor_dim0[15:0]
    g1[2] = ((k >> 16) & 0xFFFF) | (rows << 16);           // tensor_dim0[31:16] | tensor_dim1[15:0]
    g1[3] = (k << 16);                                     // tile_dim0
    g1[4] = rows;                                          // tile_dim1 (tile_dim2=0)
    g1[5] = k;                                             // tensor_dim0_stride lo
    g1[6] = 0; g1[7] = 0;
    i32x4 z4 = {0, 0, 0, 0};
#if __has_include(<hip/amd_detail/amd_gfx1250_TDM.h>)
    i32x8 z8 = {0, 0, 0, 0, 0, 0, 0, 0};
    __builtin_amdgcn_tensor_load_to_lds(g0, g1, z4, z4, z8, 0);
#else
    __builtin_amdgcn_tensor_load_to_lds(g0, g1, z4, z4, 0);
#endif
}

__device__ __forceinline__ void grid_barrier(unsigned* cnt, unsigned* rel, unsigned ep) {
    __threadfence();
    __builtin_amdgcn_s_cluster_barrier();   // HW fast-path in a cluster; S_NOP otherwise
    if (threadIdx.x == 0) {
        unsigned arrived = __hip_atomic_fetch_add(cnt, 1u, __ATOMIC_ACQ_REL, __HIP_MEMORY_SCOPE_AGENT) + 1u;
        if (arrived == ep * NWG) {
            __hip_atomic_store(rel, ep, __ATOMIC_RELEASE, __HIP_MEMORY_SCOPE_AGENT);
        } else {
            while (__hip_atomic_load(rel, __ATOMIC_ACQUIRE, __HIP_MEMORY_SCOPE_AGENT) < ep)
                __builtin_amdgcn_s_sleep(1);
        }
    }
    __syncthreads();
    __threadfence();
}

// ---------------- conversion kernels ----------------
__global__ void cvt_flat(const float* __restrict__ src, __bf16* __restrict__ dst, int n) {
    int i = blockIdx.x * blockDim.x + threadIdx.x;
    if (i < n) dst[i] = (__bf16)src[i];
}

// dst[r][c] = c < ca ? a[r][c] : b[r][c-ca]   (row-major, K = ca+cb)
__global__ void cvt_cat(const float* __restrict__ a, const float* __restrict__ b,
                        __bf16* __restrict__ dst, int rows, int ca, int cb) {
    int K = ca + cb;
    int i = blockIdx.x * blockDim.x + threadIdx.x;
    if (i >= rows * K) return;
    int r = i / K, c = i - r * K;
    float v = (c < ca) ? a[r * ca + c] : b[r * cb + (c - ca)];
    dst[i] = (__bf16)v;
}

// ---------------- persistent fused 2-layer LSTM scan ----------------
__global__ __launch_bounds__(256) void lstm_scan(
    const __bf16* __restrict__ xbf,
    const __bf16* __restrict__ w0g,       // [2048][640]  bf16
    const __bf16* __restrict__ w1g,       // [2048][1024] bf16
    const float* __restrict__ b_ih0, const float* __restrict__ b_hh0,
    const float* __restrict__ b_ih1, const float* __restrict__ b_hh1,
    __bf16* __restrict__ h0pub,           // [2][32][512] bf16 (double buffer)
    __bf16* __restrict__ h1pub,
    __bf16* __restrict__ out1,            // [B*T][512] bf16
    float* __restrict__ cOut,             // d_out + CN_OFF_EL : [2][32][512] f32
    unsigned* __restrict__ barCnt, unsigned* __restrict__ barRel)
{
    extern __shared__ char smem[];
    __bf16* W0s  = (__bf16*)(smem + LDS_W0);
    __bf16* W1s  = (__bf16*)(smem + LDS_W1);
    float*  gbuf = (float*)(smem + LDS_G);    // [gate][32][16]
    float*  c0s  = (float*)(smem + LDS_C0);   // [32][16]
    float*  c1s  = (float*)(smem + LDS_C1);
    float*  bsum = (float*)(smem + LDS_BS);   // [layer][gate][16]

    const int j    = blockIdx.x;      // which 16 h-columns
    const int tid  = threadIdx.x;
    const int w    = tid >> 5;
    const int lane = tid & 31;
    const int m    = w >> 2;          // M tile (batch half)
    const int g    = w & 3;           // gate / N tile

    // combined biases for our column slice
    if (tid < 128) {
        int l = tid >> 6, gg = (tid >> 4) & 3, c = tid & 15;
        int col = gg * H_ + j * HC + c;
        bsum[tid] = l ? (b_ih1[col] + b_hh1[col]) : (b_ih0[col] + b_hh0[col]);
    }
    // zero cell state slices
    for (int i = tid; i < 1024; i += 256) c0s[i] = 0.f;   // c0s/c1s contiguous

    // TDM: pull our 64-row weight slices (4 gate blocks x 2 layers) into LDS
    if (w == 0) {
#pragma unroll
        for (int gg = 0; gg < 4; ++gg) {
            tdm_load_2d(LDS_W0 + (unsigned)gg * HC * K0 * 2u,
                        w0g + ((size_t)(gg * H_ + j * HC)) * K0, HC, K0);
            tdm_load_2d(LDS_W1 + (unsigned)gg * HC * K1 * 2u,
                        w1g + ((size_t)(gg * H_ + j * HC)) * K1, HC, K1);
        }
        __builtin_amdgcn_s_wait_tensorcnt(0);
    }
    __syncthreads();

    const int arow = m * HC + (lane & 15);            // batch row this lane feeds (A matrix)
    const int boff = 0;                               // B fragment K select done in load_a16
    unsigned ep = 0;

#pragma unroll 1
    for (int t = 0; t < T_; ++t) {
        const int pcur = t & 1, pprev = pcur ^ 1;

        // ---------------- layer 0: gates = [x_t ; h0_{t-1}] * W0slice^T ----------------
        {
            const __bf16* xrow = xbf + ((size_t)arow * T_ + t) * DIN;
            const __bf16* hrow = h0pub + ((size_t)pprev * B_ + arow) * H_;
            const __bf16* brow = W0s + (size_t)(g * HC + (lane & 15)) * K0 + boff;
            v8f acc = {};
#pragma unroll
            for (int kc = 0; kc < K0 / 32; ++kc) {
                int kb = kc * 32;
                const __bf16* asrc = (kb < DIN) ? (xrow + kb) : (hrow + kb - DIN);
                v16bf af = load_a16(asrc, lane);
                v16bf bf = load_a16(brow + kb, lane);
                acc = __builtin_amdgcn_wmma_f32_16x16x32_bf16(false, af, false, bf,
                                                              (short)0, acc, false, false);
            }
#pragma unroll
            for (int v = 0; v < 8; ++v) {
                int rl = v + ((lane & 16) ? 8 : 0);
                gbuf[((size_t)g * 32 + m * HC + rl) * 16 + (lane & 15)] = acc[v];
            }
        }
        __syncthreads();
        // activations + state update (32 batch rows x 16 cols)
        for (int e = tid; e < 512; e += 256) {
            int row = e >> 4, col = e & 15;
            float iv = gbuf[(0 * 32 + row) * 16 + col] + bsum[0 * 64 + 0 * 16 + col];
            float fv = gbuf[(1 * 32 + row) * 16 + col] + bsum[0 * 64 + 1 * 16 + col];
            float gv = gbuf[(2 * 32 + row) * 16 + col] + bsum[0 * 64 + 2 * 16 + col];
            float ov = gbuf[(3 * 32 + row) * 16 + col] + bsum[0 * 64 + 3 * 16 + col];
            float cn = sigm(fv) * c0s[row * 16 + col] + sigm(iv) * tanh_f(gv);
            c0s[row * 16 + col] = cn;
            float hn = sigm(ov) * tanh_f(cn);
            h0pub[((size_t)pcur * B_ + row) * H_ + j * HC + col] = (__bf16)hn;
            if (t == T_ - 1) cOut[((size_t)0 * B_ + row) * H_ + j * HC + col] = cn;
        }
        grid_barrier(barCnt, barRel, ++ep);   // publish h0_t

        // ---------------- layer 1: gates = [h0_t ; h1_{t-1}] * W1slice^T ----------------
        {
            const __bf16* h0c = h0pub + ((size_t)pcur  * B_ + arow) * H_;
            const __bf16* h1p = h1pub + ((size_t)pprev * B_ + arow) * H_;
            const __bf16* brow = W1s + (size_t)(g * HC + (lane & 15)) * K1 + boff;
            v8f acc = {};
#pragma unroll
            for (int kc = 0; kc < K1 / 32; ++kc) {
                int kb = kc * 32;
                const __bf16* asrc = (kb < H_) ? (h0c + kb) : (h1p + kb - H_);
                v16bf af = load_a16(asrc, lane);
                v16bf bf = load_a16(brow + kb, lane);
                acc = __builtin_amdgcn_wmma_f32_16x16x32_bf16(false, af, false, bf,
                                                              (short)0, acc, false, false);
            }
#pragma unroll
            for (int v = 0; v < 8; ++v) {
                int rl = v + ((lane & 16) ? 8 : 0);
                gbuf[((size_t)g * 32 + m * HC + rl) * 16 + (lane & 15)] = acc[v];
            }
        }
        __syncthreads();
        for (int e = tid; e < 512; e += 256) {
            int row = e >> 4, col = e & 15;
            float iv = gbuf[(0 * 32 + row) * 16 + col] + bsum[1 * 64 + 0 * 16 + col];
            float fv = gbuf[(1 * 32 + row) * 16 + col] + bsum[1 * 64 + 1 * 16 + col];
            float gv = gbuf[(2 * 32 + row) * 16 + col] + bsum[1 * 64 + 2 * 16 + col];
            float ov = gbuf[(3 * 32 + row) * 16 + col] + bsum[1 * 64 + 3 * 16 + col];
            float cn = sigm(fv) * c1s[row * 16 + col] + sigm(iv) * tanh_f(gv);
            c1s[row * 16 + col] = cn;
            float hn = sigm(ov) * tanh_f(cn);
            h1pub[((size_t)pcur * B_ + row) * H_ + j * HC + col] = (__bf16)hn;
            out1[((size_t)row * T_ + t) * H_ + j * HC + col] = (__bf16)hn;
            if (t == T_ - 1) cOut[((size_t)1 * B_ + row) * H_ + j * HC + col] = cn;
        }
        grid_barrier(barCnt, barRel, ++ep);   // publish h1_t
    }
}

// ---------------- FC: relu(A) * fc_w^T + fc_b over out1 rows and final h states ----------------
__global__ __launch_bounds__(32) void fc_gemm(
    const __bf16* __restrict__ out1, const __bf16* __restrict__ h0pub,
    const __bf16* __restrict__ h1pub, const __bf16* __restrict__ fcw,
    const float* __restrict__ fcb, float* __restrict__ dout)
{
    const int mt = blockIdx.x;           // 0..4099 (4096 out1 tiles + 4 h_n tiles)
    const int nt = blockIdx.y;           // 0..7
    const int lane = threadIdx.x;
    const int r = mt * 16 + (lane & 15); // A row for this lane's fragment
    const int boff = 0;

    const __bf16* arow;
    if (r < B_ * T_) {
        arow = out1 + (size_t)r * H_;
    } else {
        int rr = r - B_ * T_;                         // 0..63: [h0_final ; h1_final], final parity = 1
        arow = (rr < B_) ? (h0pub + (size_t)(B_ + rr) * H_)
                         : (h1pub + (size_t)(B_ + (rr - B_)) * H_);
    }
    const __bf16* brow = fcw + (size_t)(nt * 16 + (lane & 15)) * H_ + boff;

    v8f acc = {};
#pragma unroll
    for (int kc = 0; kc < H_ / 32; ++kc) {
        v16bf af = relu16(load_a16(arow + kc * 32, lane));
        v16bf bf = load_a16(brow + kc * 32, lane);
        acc = __builtin_amdgcn_wmma_f32_16x16x32_bf16(false, af, false, bf,
                                                      (short)0, acc, false, false);
    }
    const int col = nt * 16 + (lane & 15);
    const float bias = fcb[col];
#pragma unroll
    for (int v = 0; v < 8; ++v) {
        int rr = mt * 16 + v + ((lane & 16) ? 8 : 0);
        float val = acc[v] + bias;
        if (rr < B_ * T_) dout[(size_t)rr * O_ + col] = val;                       // output [B,T,O]
        else dout[(size_t)OUT_ELEMS + (size_t)(rr - B_ * T_) * O_ + col] = val;    // hidden [2,B,O]
    }
}

// ---------------- host launcher ----------------
extern "C" void kernel_launch(void* const* d_in, const int* in_sizes, int n_in,
                              void* d_out, int out_size, void* d_ws, size_t ws_size,
                              hipStream_t stream) {
    (void)in_sizes; (void)n_in; (void)out_size; (void)ws_size;
    const float* x     = (const float*)d_in[0];
    const float* w_ih0 = (const float*)d_in[1];
    const float* w_hh0 = (const float*)d_in[2];
    const float* b_ih0 = (const float*)d_in[3];
    const float* b_hh0 = (const float*)d_in[4];
    const float* w_ih1 = (const float*)d_in[5];
    const float* w_hh1 = (const float*)d_in[6];
    const float* b_ih1 = (const float*)d_in[7];
    const float* b_hh1 = (const float*)d_in[8];
    const float* fc_w  = (const float*)d_in[9];
    const float* fc_b  = (const float*)d_in[10];

    char* ws = (char*)d_ws;
    __bf16*   xbf   = (__bf16*)(ws + X_OFF);
    __bf16*   w0cat = (__bf16*)(ws + W0_OFF);
    __bf16*   w1cat = (__bf16*)(ws + W1_OFF);
    __bf16*   fcbf  = (__bf16*)(ws + FC_OFF);
    __bf16*   h0pub = (__bf16*)(ws + H0_OFF);
    __bf16*   h1pub = (__bf16*)(ws + H1_OFF);
    __bf16*   out1  = (__bf16*)(ws + OUT1_OFF);
    unsigned* bar   = (unsigned*)(ws + BAR_OFF);

    hipFuncSetAttribute(reinterpret_cast<const void*>(lstm_scan),
                        hipFuncAttributeMaxDynamicSharedMemorySize, (int)SMEM_BYTES);

    // precision conversion / weight concatenation (bf16, L2-resident working set)
    cvt_flat<<<(B_ * T_ * DIN + 255) / 256, 256, 0, stream>>>(x, xbf, B_ * T_ * DIN);
    cvt_cat<<<(G4 * K0 + 255) / 256, 256, 0, stream>>>(w_ih0, w_hh0, w0cat, G4, DIN, H_);
    cvt_cat<<<(G4 * K1 + 255) / 256, 256, 0, stream>>>(w_ih1, w_hh1, w1cat, G4, H_, H_);
    cvt_flat<<<(O_ * H_ + 255) / 256, 256, 0, stream>>>(fc_w, fcbf, O_ * H_);

    // zero initial h states and the grid barrier (graph-replay deterministic)
    hipMemsetAsync(h0pub, 0, HP_BYTES, stream);
    hipMemsetAsync(h1pub, 0, HP_BYTES, stream);
    hipMemsetAsync(bar, 0, 256, stream);

    float* dout = (float*)d_out;
    lstm_scan<<<NWG, 256, SMEM_BYTES, stream>>>(
        xbf, w0cat, w1cat, b_ih0, b_hh0, b_ih1, b_hh1,
        h0pub, h1pub, out1, dout + CN_OFF_EL, bar, bar + 1);

    fc_gemm<<<dim3((B_ * T_ + 2 * B_) / 16, O_ / 16), 32, 0, stream>>>(
        out1, h0pub, h1pub, fcbf, fc_b, dout);
}
